// SCPA_rep2_77876347011783
// MI455X (gfx1250) — matrix operand order repaired and checked by reading, not compiled, not run on hardware
//
#include <hip/hip_runtime.h>
#include <hip/hip_bf16.h>

#define HH  128
#define WW2 128
#define CHW (HH*WW2)       // 16384
#define BB  8
#define PP  (BB*CHW)       // 131072 pixels

typedef __attribute__((ext_vector_type(16))) _Float16 v16h;
typedef __attribute__((ext_vector_type(8)))  float    v8f;

union AFrag { v16h v; _Float16 e[16]; };
union CFrag { v8f  v; float    e[8];  };

__device__ __forceinline__ float lrelu_f(float x){ return x >= 0.f ? x : 0.2f*x; }
__device__ __forceinline__ float sigm_f (float x){ return 1.f/(1.f + expf(-x)); }

// ---------------------------------------------------------------------------
// Stage A: expand 1-channel condition into flows and 1-in-channel conv outputs.
// GEMM-only consumers -> store activations as f16 (halves HBM traffic).
// ---------------------------------------------------------------------------
__global__ __launch_bounds__(256)
void k_stage_a(const float* __restrict__ cond,
               const float* __restrict__ c1w,  const float* __restrict__ c1b,
               const float* __restrict__ pacw, const float* __restrict__ pacb,
               const float* __restrict__ o2w1, const float* __restrict__ o2b1,
               const float* __restrict__ o3w1, const float* __restrict__ o3b1,
               const float* __restrict__ paw1, const float* __restrict__ pab1,
               float* __restrict__ fx,  float* __restrict__ fy,
               float* __restrict__ pfx, float* __restrict__ pfy,
               _Float16* __restrict__ t2a, _Float16* __restrict__ t3a,
               _Float16* __restrict__ pat)
{
    int p = blockIdx.x*256 + threadIdx.x;
    float c = cond[p];
    fx [p] = lrelu_f(c*c1w [0] + c1b [0]);  // flow channel 0 -> x
    fy [p] = lrelu_f(c*c1w [1] + c1b [1]);  // flow channel 1 -> y
    pfx[p] = lrelu_f(c*pacw[0] + pacb[0]);
    pfy[p] = lrelu_f(c*pacw[1] + pacb[1]);
    int b = p >> 14, hw = p & (CHW-1);
    _Float16* t2p = t2a + (size_t)b*128*CHW + hw;
    for (int k = 0; k < 128; ++k) t2p[(size_t)k*CHW] = (_Float16)lrelu_f(c*o2w1[k] + o2b1[k]);
    _Float16* t3p = t3a + (size_t)b*256*CHW + hw;
    for (int k = 0; k < 256; ++k) t3p[(size_t)k*CHW] = (_Float16)lrelu_f(c*o3w1[k] + o3b1[k]);
    _Float16* pp  = pat + (size_t)b*64*CHW + hw;
    for (int k = 0; k < 64;  ++k) pp[(size_t)k*CHW]  = (_Float16)lrelu_f(c*paw1[k] + pab1[k]);
}

// ---------------------------------------------------------------------------
// Generic 1x1-conv GEMM: D[b,o,hw] = act(sum_k W[o,k]*S[b,k,hw] + bias[o])
// Block: 256 thr = 8 waves; tile 32(O) x 64(HW); f16 WMMA, f32 accumulate.
// - K / input type / output type are template params.
// - f16 input: one uint4 (8 halves) per thread per K-step, no cvts.
// - Activation tile TRANSPOSED in LDS ([n][k], 40-half rows): B fragment is
//   16 contiguous halves -> 2x ds_load_b128. Weight rows padded to K+8.
// - Double-buffered LDS tile, global->reg prefetch, one barrier per K-step.
// ---------------------------------------------------------------------------
template<int K, typename TIN, typename TOUT>
__global__ __launch_bounds__(256)
void k_gemm_wmma(const float* __restrict__ Wt, const float* __restrict__ bias,
                 const TIN* __restrict__ S,  TOUT* __restrict__ D,
                 int O, int act)
{
    constexpr bool IN16 = (sizeof(TIN) == 2);
    constexpr int KP = K + 8;           // padded weight row (halves)
    constexpr int NS = 40;              // padded tile row (32 k + 8 pad)
    __shared__ _Float16 wlds[32*KP];
    __shared__ _Float16 stile[2][64*NS];
    const int tid   = threadIdx.x;
    const int b     = blockIdx.z;
    const int hw0   = blockIdx.x*64;
    const int obase = blockIdx.y*32;

    for (int idx = tid; idx < 32*K; idx += 256) {
        int r = idx / K, k = idx - r*K;
        wlds[r*KP + k] = (_Float16)Wt[(size_t)(obase + r)*K + k];
    }

    const TIN* Sb = S + (size_t)b*K*CHW + hw0;
    float4 pf[2]; uint4 ph;
    auto gload = [&](int ks) {
        if constexpr (IN16) {
            int kk = tid >> 3, n8 = tid & 7;
            ph = *reinterpret_cast<const uint4*>(
                     reinterpret_cast<const _Float16*>(Sb) + (size_t)(ks+kk)*CHW + n8*8);
        } else {
            #pragma unroll
            for (int i = 0; i < 2; ++i) {
                int e = tid + i*256, kk = e >> 4, n4 = e & 15;
                pf[i] = *reinterpret_cast<const float4*>(
                            reinterpret_cast<const float*>(Sb) + (size_t)(ks+kk)*CHW + n4*4);
            }
        }
    };
    auto lstore = [&](int bufi) {
        if constexpr (IN16) {
            int kk = tid >> 3, n8 = tid & 7;
            union { uint4 u; _Float16 h[8]; } v; v.u = ph;
            #pragma unroll
            for (int q = 0; q < 8; ++q)
                stile[bufi][(n8*8+q)*NS + kk] = v.h[q];
        } else {
            #pragma unroll
            for (int i = 0; i < 2; ++i) {
                int e = tid + i*256, kk = e >> 4, n4 = e & 15;
                stile[bufi][(n4*4+0)*NS + kk] = (_Float16)pf[i].x;
                stile[bufi][(n4*4+1)*NS + kk] = (_Float16)pf[i].y;
                stile[bufi][(n4*4+2)*NS + kk] = (_Float16)pf[i].z;
                stile[bufi][(n4*4+3)*NS + kk] = (_Float16)pf[i].w;
            }
        }
    };

    gload(0);
    lstore(0);

    const int lane = tid & 31, ln = lane & 15, hi = lane >> 4;
    const int wv = tid >> 5, mt = wv >> 2, nt = wv & 3;
    CFrag acc;
    #pragma unroll
    for (int r = 0; r < 8; ++r) acc.e[r] = 0.f;

    __syncthreads();
    int buf = 0;
    for (int ks = 0; ks < K; ks += 32) {
        const bool more = (ks + 32) < K;
        if (more) gload(ks + 32);
        AFrag a, bfr;
        #pragma unroll
        for (int j = 0; j < 16; ++j) {
            int ka = hi*16 + j;                            // A: contiguous
            a.e[j]   = wlds[(mt*16 + ln)*KP + ks + ka];
            bfr.e[j] = stile[buf][(nt*16 + ln)*NS + hi*16 + j];  // B: contiguous
        }
        acc.v = __builtin_amdgcn_wmma_f32_16x16x32_f16(false, a.v, false, bfr.v,
                                                       (short)0, acc.v, false, false);
        if (more) lstore(buf ^ 1);
        __syncthreads();
        buf ^= 1;
    }
    #pragma unroll
    for (int r = 0; r < 8; ++r) {
        int row = obase + mt*16 + hi*8 + r;
        float v = acc.e[r];
        if (bias) v += bias[row];
        if (act == 1) v = lrelu_f(v);
        D[(size_t)b*O*CHW + (size_t)row*CHW + hw0 + nt*16 + ln] = (TOUT)v;
    }
}

// ---------------------------------------------------------------------------
// DCN-1x1: bilinear gather (border=False) * sigmoid-mask, then O=64 x C GEMM
// Gathered tile stored transposed [pixel][c] (C+8 padded) for b128 B-frags.
// ---------------------------------------------------------------------------
template<int C>
__global__ __launch_bounds__(256)
void k_dcn_wmma(const float* __restrict__ src, const float* __restrict__ dy,
                const float* __restrict__ dx,  const float* __restrict__ m,
                const float* __restrict__ Wt,  const float* __restrict__ bias,
                float* __restrict__ out)
{
    constexpr int CP = C + 8;
    __shared__ _Float16 wlds[64*CP];
    __shared__ _Float16 slds[64*CP];     // [px][c]
    __shared__ int   py0[64], px0[64];
    __shared__ float pwy[64], pwx[64], pmm[64];
    const int tid = threadIdx.x;
    const int b   = blockIdx.z;
    const int hw0 = blockIdx.x*64;
    for (int idx = tid; idx < 64*C; idx += 256) {
        int r = idx / C, k = idx - r*C;
        wlds[r*CP + k] = (_Float16)Wt[idx];
    }
    if (tid < 64) {
        int hw = hw0 + tid;
        int p  = b*CHW + hw;
        float ys = (float)(hw >> 7) + dy[p];
        float xs = (float)(hw & 127) + dx[p];
        float y0f = floorf(ys), x0f = floorf(xs);
        py0[tid] = (int)y0f;  px0[tid] = (int)x0f;
        pwy[tid] = ys - y0f;  pwx[tid] = xs - x0f;
        pmm[tid] = m[p];
    }
    __syncthreads();
    for (int idx = tid; idx < C*64; idx += 256) {
        int c = idx >> 6, px = idx & 63;
        int y0 = py0[px], x0 = px0[px];
        float wy = pwy[px], wx = pwx[px];
        const float* sp = src + (size_t)b*C*CHW + (size_t)c*CHW;
        int y1 = y0 + 1, x1 = x0 + 1;
        bool yv0 = (y0 >= 0) & (y0 < HH),  yv1 = (y1 >= 0) & (y1 < HH);
        bool xv0 = (x0 >= 0) & (x0 < WW2), xv1 = (x1 >= 0) & (x1 < WW2);
        float v = 0.f;
        if (yv0 & xv0) v += sp[y0*WW2 + x0] * (1.f-wy)*(1.f-wx);
        if (yv0 & xv1) v += sp[y0*WW2 + x1] * (1.f-wy)*wx;
        if (yv1 & xv0) v += sp[y1*WW2 + x0] * wy*(1.f-wx);
        if (yv1 & xv1) v += sp[y1*WW2 + x1] * wy*wx;
        slds[px*CP + c] = (_Float16)(v * pmm[px]);
    }
    __syncthreads();
    const int lane = tid & 31, ln = lane & 15, hi = lane >> 4, wv = tid >> 5;
    #pragma unroll
    for (int t0 = 0; t0 < 2; ++t0) {
        int t = wv + t0*8;
        int mt = t >> 2, nt = t & 3;
        CFrag acc;
        #pragma unroll
        for (int r = 0; r < 8; ++r) acc.e[r] = 0.f;
        #pragma unroll
        for (int ks = 0; ks < C; ks += 32) {
            AFrag a, bfr;
            #pragma unroll
            for (int j = 0; j < 16; ++j) {
                int ka = hi*16 + j;
                a.e[j]   = wlds[(mt*16 + ln)*CP + ks + ka];
                bfr.e[j] = slds[(nt*16 + ln)*CP + ks + hi*16 + j];
            }
            acc.v = __builtin_amdgcn_wmma_f32_16x16x32_f16(false, a.v, false, bfr.v,
                                                           (short)0, acc.v, false, false);
        }
        #pragma unroll
        for (int r = 0; r < 8; ++r) {
            int row = mt*16 + hi*8 + r;
            out[(size_t)b*64*CHW + (size_t)row*CHW + hw0 + nt*16 + ln] = acc.e[r] + bias[row];
        }
    }
}

// ---------------------------------------------------------------------------
// 3-channel offset conv (f16 features in): dy, dx, m=sigmoid
// ---------------------------------------------------------------------------
template<int K>
__global__ __launch_bounds__(256)
void k_off3(const float* __restrict__ ow, const float* __restrict__ ob,
            const _Float16* __restrict__ feat, float* __restrict__ dy,
            float* __restrict__ dx, float* __restrict__ m)
{
    int p = blockIdx.x*256 + threadIdx.x;
    int b = p >> 14, hw = p & (CHW-1);
    const _Float16* fp = feat + (size_t)b*K*CHW + hw;
    float a0 = 0.f, a1 = 0.f, a2 = 0.f;
    #pragma unroll 4
    for (int k = 0; k < K; ++k) {
        float v = (float)fp[(size_t)k*CHW];
        a0 += v*ow[k]; a1 += v*ow[K+k]; a2 += v*ow[2*K+k];
    }
    dy[p] = a0 + ob[0];
    dx[p] = a1 + ob[1];
    m [p] = sigm_f(a2 + ob[2]);
}

// ---------------------------------------------------------------------------
// t = act(t + flow_warp(t, flow))   (border=True clamp), C = 64
// ---------------------------------------------------------------------------
__global__ __launch_bounds__(256)
void k_warp_add(const float* __restrict__ src, const float* __restrict__ fx,
                const float* __restrict__ fy, float* __restrict__ dst, int act)
{
    int idx = blockIdx.x*256 + threadIdx.x;   // BB*64*CHW threads
    int b = idx / (64*CHW);
    int rem = idx - b*64*CHW;
    int hw = rem & (CHW-1);
    int p = b*CHW + hw;
    float ys = (float)(hw >> 7) + fy[p];
    float xs = (float)(hw & 127) + fx[p];
    ys = fminf(fmaxf(ys, 0.f), (float)(HH-1));
    xs = fminf(fmaxf(xs, 0.f), (float)(WW2-1));
    float y0f = floorf(ys), x0f = floorf(xs);
    int y0 = (int)y0f, x0 = (int)x0f;
    float wy = ys - y0f, wx = xs - x0f;
    int y1 = min(y0+1, HH-1), x1 = min(x0+1, WW2-1);
    int ch = rem >> 14;
    const float* sp = src + (size_t)b*64*CHW + (size_t)ch*CHW;
    float v = sp[y0*WW2+x0]*(1.f-wy)*(1.f-wx) + sp[y0*WW2+x1]*(1.f-wy)*wx
            + sp[y1*WW2+x0]*wy*(1.f-wx)       + sp[y1*WW2+x1]*wy*wx;
    float r = src[idx] + v;
    dst[idx] = (act == 0) ? lrelu_f(r) : sigm_f(r);
}

// ---------------------------------------------------------------------------
// Depthwise 3x3 (pad 1) with optional bias / elementwise multiply / residual
// ---------------------------------------------------------------------------
__global__ __launch_bounds__(256)
void k_dw3x3(const float* __restrict__ src, const float* __restrict__ kern,
             const float* __restrict__ bias, const float* __restrict__ mul,
             const float* __restrict__ resid, float* __restrict__ dst, int C)
{
    int idx = blockIdx.x*256 + threadIdx.x;
    int b = idx / (C*CHW);
    int rem = idx - b*C*CHW;
    int ch = rem / CHW;
    int hw = rem - ch*CHW;
    int row = hw >> 7, col = hw & 127;
    const float* sp = src + (size_t)b*C*CHW + (size_t)ch*CHW;
    const float* kp = kern + ch*9;
    float s = 0.f;
    #pragma unroll
    for (int dr = -1; dr <= 1; ++dr)
        #pragma unroll
        for (int dc = -1; dc <= 1; ++dc) {
            int rr = row + dr, cc = col + dc;
            if (rr >= 0 && rr < HH && cc >= 0 && cc < WW2)
                s += sp[rr*WW2 + cc] * kp[(dr+1)*3 + (dc+1)];
        }
    if (bias)  s += bias[ch];
    if (mul)   s *= mul[idx];
    if (resid) s += resid[idx];
    dst[idx] = s;
}

// ---------------------------------------------------------------------------
// concat(lrelu(a), lrelu(b)) into 128 f16 channels (GEMM-only consumer)
// ---------------------------------------------------------------------------
__global__ __launch_bounds__(256)
void k_cat_lrelu(const float* __restrict__ a, const float* __restrict__ b2,
                 _Float16* __restrict__ dst)
{
    int idx = blockIdx.x*256 + threadIdx.x;  // BB*128*CHW
    int b = idx / (128*CHW);
    int rem = idx - b*128*CHW;
    int ch = rem >> 14;
    int hw = rem & (CHW-1);
    float v = (ch < 64) ? a [(size_t)b*64*CHW + (size_t)ch*CHW + hw]
                        : b2[(size_t)b*64*CHW + (size_t)(ch-64)*CHW + hw];
    dst[idx] = (_Float16)lrelu_f(v);
}

// ---------------------------------------------------------------------------
// Fused avgpool3 + FAM + concat:
//   x_a = 3x3 zero-pad mean(out); fam = x_a*(p3+p4) + out*(p1-p4)
//   cc  = concat(out, fam) as f16 (GEMM-only consumer)
// ---------------------------------------------------------------------------
__global__ __launch_bounds__(256)
void k_famcc(const float* __restrict__ out,
             const float* __restrict__ p1, const float* __restrict__ p2,
             const float* __restrict__ p3, const float* __restrict__ p4,
             _Float16* __restrict__ cc)
{
    int idx = blockIdx.x*256 + threadIdx.x;  // BB*128*CHW
    int b = idx / (128*CHW);
    int rem = idx - b*128*CHW;
    int ch = rem >> 14;
    int hw = rem & (CHW-1);
    int row = hw >> 7, col = hw & 127;
    const float* sp = out + (size_t)b*128*CHW + (size_t)ch*CHW;
    float s = 0.f;
    #pragma unroll
    for (int dr = -1; dr <= 1; ++dr)
        #pragma unroll
        for (int dc = -1; dc <= 1; ++dc) {
            int rr = row + dr, c2 = col + dc;
            if (rr >= 0 && rr < HH && c2 >= 0 && c2 < WW2)
                s += sp[rr*WW2 + c2];
        }
    float xa = s * (1.f/9.f);
    float o  = sp[hw & (CHW-1) - 0 + 0*CHW + (row*WW2 + col) - (row*WW2 + col)]; // = sp[row*WW2+col]
    o = sp[row*WW2 + col];
    float fam = xa*(p3[0] + p4[0]) + o*(p1[0] - p4[0]);   // p2 term is zero
    cc[(size_t)b*256*CHW + (size_t)ch*CHW + hw]       = (_Float16)o;
    cc[(size_t)b*256*CHW + (size_t)(128+ch)*CHW + hw] = (_Float16)fam;
    (void)p2;
}

// ---------------------------------------------------------------------------
extern "C" void kernel_launch(void* const* d_in, const int* in_sizes, int n_in,
                              void* d_out, int out_size, void* d_ws, size_t ws_size,
                              hipStream_t stream)
{
    const float* x          = (const float*)d_in[0];
    const float* cond       = (const float*)d_in[1];
    const float* cond1_w    = (const float*)d_in[2];
    const float* cond1_b    = (const float*)d_in[3];
    const float* oc2_w1     = (const float*)d_in[4];
    const float* oc2_b1     = (const float*)d_in[5];
    const float* oc2_w2     = (const float*)d_in[6];
    const float* oc2_b2     = (const float*)d_in[7];
    const float* dcn2_off_w = (const float*)d_in[8];
    const float* dcn2_off_b = (const float*)d_in[9];
    const float* dcn2_w     = (const float*)d_in[10];
    const float* dcn2_b     = (const float*)d_in[11];
    const float* oc3_w1     = (const float*)d_in[12];
    const float* oc3_b1     = (const float*)d_in[13];
    const float* oc3_w2     = (const float*)d_in[14];
    const float* oc3_b2     = (const float*)d_in[15];
    const float* dcn3_off_w = (const float*)d_in[16];
    const float* dcn3_off_b = (const float*)d_in[17];
    const float* dcn3_w     = (const float*)d_in[18];
    const float* dcn3_b     = (const float*)d_in[19];
    const float* k1_pw      = (const float*)d_in[20];
    const float* k1_dw      = (const float*)d_in[21];
    const float* k1_db      = (const float*)d_in[22];
    const float* pa_cond_w  = (const float*)d_in[23];
    const float* pa_cond_b  = (const float*)d_in[24];
    const float* pa_oc_w1   = (const float*)d_in[25];
    const float* pa_oc_b1   = (const float*)d_in[26];
    const float* pa_oc_w2   = (const float*)d_in[27];
    const float* pa_oc_b2   = (const float*)d_in[28];
    const float* pa_off_w   = (const float*)d_in[29];
    const float* pa_off_b   = (const float*)d_in[30];
    const float* pa_dcn_w   = (const float*)d_in[31];
    const float* pa_dcn_b   = (const float*)d_in[32];
    const float* pa_k3_pw   = (const float*)d_in[33];
    const float* pa_k3_dw   = (const float*)d_in[34];
    const float* pa_k4_pw   = (const float*)d_in[35];
    const float* pa_k4_dw   = (const float*)d_in[36];
    const float* conv3_w    = (const float*)d_in[37];
    const float* fam_p1     = (const float*)d_in[38];
    const float* fam_p2     = (const float*)d_in[39];
    const float* fam_p3     = (const float*)d_in[40];
    const float* fam_p4     = (const float*)d_in[41];
    const float* co_pw      = (const float*)d_in[42];
    const float* co_dw      = (const float*)d_in[43];
    const float* co_db      = (const float*)d_in[44];

    float* ws = (float*)d_ws;
    const size_t P = (size_t)PP;
    float* flow_x = ws + 0*P;   float* flow_y = ws + 1*P;
    float* pflx   = ws + 2*P;   float* pfly   = ws + 3*P;
    float* dy2 = ws + 4*P;  float* dx2 = ws + 5*P;  float* m2  = ws + 6*P;
    float* dy3 = ws + 7*P;  float* dx3 = ws + 8*P;  float* m3  = ws + 9*P;
    float* pdy = ws + 10*P; float* pdx = ws + 11*P; float* pm  = ws + 12*P;
    _Float16* t3a16  = (_Float16*)(ws + 13*P);   // 256P halves ; reused as cc16
    _Float16* t2a16  = (_Float16*)(ws + 269*P);  // 128P halves ; reused as cat1
    _Float16* pat16  = (_Float16*)(ws + 397*P);  //  64P halves
    _Float16* feat16 = (_Float16*)(ws + 461*P);  // 128P halves ; region reused as f32 'out'
    float* B0 = ws + 589*P; float* B1 = ws + 653*P; float* B2 = ws + 717*P;
    float* B3 = ws + 781*P; float* B4 = ws + 845*P; float* B5 = ws + 909*P;
    float* r1 = ws + 973*P;     // 128P
    _Float16* cat116 = t2a16; _Float16* cc16 = t3a16;
    float* outb = ws + 461*P;   // f32, after feat16 is dead

    const int nblkP  = PP/256;                // per-pixel kernels
    const int nblk64 = (BB*64*CHW)/256;       // 64-channel elementwise
    const int nblk128= (BB*128*CHW)/256;      // 128-channel elementwise
    dim3 blk(256);
    dim3 g64 (CHW/64, 64/32,  BB);            // GEMM grids
    dim3 g128(CHW/64, 128/32, BB);
    dim3 gdcn(CHW/64, 1, BB);

    // 1. condition expansions (f16 activations)
    k_stage_a<<<nblkP, blk, 0, stream>>>(cond, cond1_w, cond1_b, pa_cond_w, pa_cond_b,
        oc2_w1, oc2_b1, oc3_w1, oc3_b1, pa_oc_w1, pa_oc_b1,
        flow_x, flow_y, pflx, pfly, t2a16, t3a16, pat16);
    // 2-7. feature GEMMs (f16 in / f16 out) + DCN offsets
    k_gemm_wmma<128, _Float16, _Float16><<<g128, blk, 0, stream>>>(oc2_w2, oc2_b2, t2a16, feat16, 128, 1);
    k_off3<128><<<nblkP, blk, 0, stream>>>(dcn2_off_w, dcn2_off_b, feat16, dy2, dx2, m2);
    k_gemm_wmma<256, _Float16, _Float16><<<g128, blk, 0, stream>>>(oc3_w2, oc3_b2, t3a16, feat16, 128, 1);
    k_off3<128><<<nblkP, blk, 0, stream>>>(dcn3_off_w, dcn3_off_b, feat16, dy3, dx3, m3);
    k_gemm_wmma<64, _Float16, _Float16><<<g64, blk, 0, stream>>>(pa_oc_w2, pa_oc_b2, pat16, feat16, 64, 1);
    k_off3<64><<<nblkP, blk, 0, stream>>>(pa_off_w, pa_off_b, feat16, pdy, pdx, pm);
    // 8-9. DCN gathers + group GEMM
    k_dcn_wmma<128><<<gdcn, blk, 0, stream>>>(x, dy2, dx2, m2, dcn2_w, dcn2_b, B0);
    k_dcn_wmma<128><<<gdcn, blk, 0, stream>>>(x, dy3, dx3, m3, dcn3_w, dcn3_b, B1);
    // 10-11. warp + add + lrelu
    k_warp_add<<<nblk64, blk, 0, stream>>>(B0, flow_x, flow_y, B2, 0);
    k_warp_add<<<nblk64, blk, 0, stream>>>(B1, flow_x, flow_y, B3, 0);
    // 12. out_a = bsconv(out_a, k1)
    k_gemm_wmma<64, float, float><<<g64, blk, 0, stream>>>(k1_pw, nullptr, B2, B4, 64, 0);
    k_dw3x3<<<nblk64, blk, 0, stream>>>(B4, k1_dw, k1_db, nullptr, nullptr, B0, 64);
    // 13-14. PA branch: y = sigmoid(dcn(out_b) + warp)
    k_dcn_wmma<64><<<gdcn, blk, 0, stream>>>(B3, pdy, pdx, pm, pa_dcn_w, pa_dcn_b, B5);
    k_warp_add<<<nblk64, blk, 0, stream>>>(B5, pflx, pfly, B1, 1);
    // 15-16. out_b = bsconv(bsconv(out_b,k3)*y, k4)
    k_gemm_wmma<64, float, float><<<g64, blk, 0, stream>>>(pa_k3_pw, nullptr, B3, B5, 64, 0);
    k_dw3x3<<<nblk64, blk, 0, stream>>>(B5, pa_k3_dw, nullptr, B1, nullptr, B2, 64);
    k_gemm_wmma<64, float, float><<<g64, blk, 0, stream>>>(pa_k4_pw, nullptr, B2, B4, 64, 0);
    k_dw3x3<<<nblk64, blk, 0, stream>>>(B4, pa_k4_dw, nullptr, nullptr, nullptr, B3, 64);
    // 17-18. out = conv3(concat(lrelu(out_a), lrelu(out_b)))
    k_cat_lrelu<<<nblk128, blk, 0, stream>>>(B0, B3, cat116);
    k_gemm_wmma<128, _Float16, float><<<g128, blk, 0, stream>>>(conv3_w, nullptr, cat116, outb, 128, 0);
    // 19. fused avgpool + FAM + concat (f16)
    k_famcc<<<nblk128, blk, 0, stream>>>(outb, fam_p1, fam_p2, fam_p3, fam_p4, cc16);
    // 20-21. res = bsconv(cc, co) + x  -> d_out
    k_gemm_wmma<256, _Float16, float><<<g128, blk, 0, stream>>>(co_pw, nullptr, cc16, r1, 128, 0);
    k_dw3x3<<<nblk128, blk, 0, stream>>>(r1, co_dw, co_db, nullptr, x, (float*)d_out, 128);

    (void)in_sizes; (void)n_in; (void)out_size; (void)ws_size;
}
